// Gnn_53120155517255
// MI455X (gfx1250) — matrix operand order repaired and checked by reading, not compiled, not run on hardware
//
#include <hip/hip_runtime.h>

#define D1 384
#define D2 768
#define LEAKY 0.01f

typedef __attribute__((ext_vector_type(16))) __bf16 v16bf;
typedef __attribute__((ext_vector_type(8)))  float  v8f;

static __device__ __forceinline__ unsigned short f2bf(float f) {
    unsigned int u = __float_as_uint(f);
    u += 0x7FFFu + ((u >> 16) & 1u);   // round-to-nearest-even
    return (unsigned short)(u >> 16);
}
static __device__ __forceinline__ unsigned int pk2(float a, float b) {
    return (unsigned int)f2bf(a) | ((unsigned int)f2bf(b) << 16);
}
static __device__ __forceinline__ float lrelu(float v) {
    return v >= 0.0f ? v : LEAKY * v;
}

// ---------------------------------------------------------------------------
// Degree / normalization
// ---------------------------------------------------------------------------
__global__ void k_fill1(float* p, int n) {
    int i = blockIdx.x * blockDim.x + threadIdx.x;
    if (i < n) p[i] = 1.0f;                      // self-loop contributes 1
}
__global__ void k_deg(const int* __restrict__ dst, float* deg, int E) {
    int e = blockIdx.x * blockDim.x + threadIdx.x;
    if (e < E) atomicAdd(&deg[dst[e]], 1.0f);
}
__global__ void k_rsqrt(float* p, int n) {
    int i = blockIdx.x * blockDim.x + threadIdx.x;
    if (i < n) p[i] = rsqrtf(p[i]);              // deg >= 1 always
}

// ---------------------------------------------------------------------------
// Pre-swizzle weights [K x 384] f32 -> bf16 in WMMA B-fragment layout:
//   Wsw[((kt*24 + nt)*32 + lane)*16 + e],  k = kt*32 + (lane/16)*16 + e,
//                                          n = nt*16 + (lane%16)
// Each lane's 16 elements are 32 contiguous bytes -> b128 loads everywhere.
// ---------------------------------------------------------------------------
__global__ void k_wswz(const float* __restrict__ W, unsigned short* __restrict__ Wsw,
                       int K) {
    int o = blockIdx.x * blockDim.x + threadIdx.x;
    if (o >= K * D1) return;
    int e  = o & 15;
    int l  = (o >> 4) & 31;
    int nt = (o >> 9) % 24;
    int kt = o / (24 * 512);
    int k  = kt * 32 + (l >> 4) * 16 + e;
    int n  = nt * 16 + (l & 15);
    Wsw[o] = f2bf(W[(size_t)k * D1 + n]);
}

// ---------------------------------------------------------------------------
// WMMA GEMM:  C[M x 384] = A[M x K] (f32 or bf16) * W (pre-swizzled bf16)
// Block tile 128(M) x 64(N), BK = 32, 256 threads = 8 waves, wave = 32x32.
// Dynamic LDS: [B panel: KT*4 tiles fragment-major][A tile: 8 tiles frag-major]
// ---------------------------------------------------------------------------
union Frag { uint4 q[2]; v16bf v; };

__global__ __launch_bounds__(256)
void k_gemm_bf16(const void* __restrict__ Ap, int aIsBf16, int lda,
                 const unsigned short* __restrict__ Wsw,
                 float* __restrict__ C, int M, int K) {
    extern __shared__ unsigned short smem[];
    const int KT = K >> 5;
    unsigned short* Bpan = smem;                  // KT*4*32*16 bf16
    unsigned short* Apan = smem + (size_t)KT * 2048;  // 8*32*16 bf16

    const int t     = threadIdx.x;
    const int lane  = t & 31;
    const int wid   = t >> 5;
    const int wm    = wid & 3;          // 4 waves along M (32 rows each)
    const int wn    = wid >> 2;         // 2 waves along N (32 cols each)
    const int mBase = blockIdx.x * 128;
    const int nb0   = blockIdx.y * 4;   // base n-tile (of 16 cols)

    // ---- stage entire B panel (K x 64) once: pure b128 copies ----
    {
        const uint4* Wv = (const uint4*)Wsw;
        uint4*       Bv = (uint4*)Bpan;
        int totB = KT * 4 * 64;                     // uint4 chunks
        for (int idx = t; idx < totB; idx += 256) {
            int q   = idx & 63;
            int ntL = (idx >> 6) & 3;
            int kt  = idx >> 8;
            Bv[idx] = Wv[(size_t)((kt * 24 + nb0 + ntL) << 6) + q];
        }
    }

    v8f acc[2][2] = {};
    uint4* Aw = (uint4*)Apan;

    for (int kt = 0; kt < KT; ++kt) {
        __syncthreads();
        // ---- stage A tile (128 x 32) fragment-major ----
        #pragma unroll
        for (int i = 0; i < 2; ++i) {
            int q  = t * 2 + i;                    // 512 chunks of 8 k-values
            int m  = q >> 2, g = q & 3;            // g: 8-k group within row
            int rg = mBase + m;
            uint4 u = make_uint4(0, 0, 0, 0);
            if (aIsBf16) {
                if (rg < M)
                    u = *(const uint4*)((const unsigned short*)Ap +
                                        (size_t)rg * lda + kt * 32 + g * 8);
            } else {
                if (rg < M) {
                    const float4* p = (const float4*)((const float*)Ap +
                                      (size_t)rg * lda + kt * 32 + g * 8);
                    float4 v0 = p[0], v1 = p[1];
                    u = make_uint4(pk2(v0.x, v0.y), pk2(v0.z, v0.w),
                                   pk2(v1.x, v1.y), pk2(v1.z, v1.w));
                }
            }
            // dst lane/reg-half: contiguous k order (k = 16*half + s), matching B
            int dl = (m & 15) + (g >> 1) * 16;
            Aw[(size_t)((m >> 4) * 32 + dl) * 2 + (g & 1)] = u;
        }
        __syncthreads();

        // ---- load fragments: 2x b128 each, contiguous per lane ----
        Frag af[2], bfr[2];
        #pragma unroll
        for (int i = 0; i < 2; ++i) {
            const uint4* p = Aw + (size_t)(((wm * 2 + i) * 32 + lane)) * 2;
            af[i].q[0] = p[0]; af[i].q[1] = p[1];
        }
        const uint4* Bq = (const uint4*)Bpan;
        #pragma unroll
        for (int j = 0; j < 2; ++j) {
            const uint4* p = Bq + (size_t)(((kt * 4 + wn * 2 + j) * 32 + lane)) * 2;
            bfr[j].q[0] = p[0]; bfr[j].q[1] = p[1];
        }
        #pragma unroll
        for (int i = 0; i < 2; ++i)
            #pragma unroll
            for (int j = 0; j < 2; ++j)
                acc[i][j] = __builtin_amdgcn_wmma_f32_16x16x32_bf16(
                    false, af[i].v, false, bfr[j].v,
                    (short)0, acc[i][j], false, false);
    }

    // ---- store C (f32 C/D layout: lane half*8+r -> M, lane%16 -> N) ----
    const int mrow = lane & 15, half = lane >> 4;
    #pragma unroll
    for (int i = 0; i < 2; ++i)
        #pragma unroll
        for (int j = 0; j < 2; ++j)
            #pragma unroll
            for (int r = 0; r < 8; ++r) {
                int m = mBase + wm * 32 + i * 16 + half * 8 + r;
                int n = nb0 * 16 + wn * 32 + j * 16 + mrow;
                if (m < M) C[(size_t)m * D1 + n] = acc[i][j][r];
            }
}

// ---------------------------------------------------------------------------
// Aggregation: tmp = dinv^2 * h (self loop), then scatter-add over edges
// ---------------------------------------------------------------------------
__global__ void k_selfinit(const float* __restrict__ h, const float* __restrict__ dinv,
                           float* __restrict__ tmp, int total4) {   // N*96
    int i = blockIdx.x * blockDim.x + threadIdx.x;
    if (i >= total4) return;
    int node = i / 96;
    float s = dinv[node]; s *= s;
    float4 v = ((const float4*)h)[i];
    ((float4*)tmp)[i] = make_float4(s * v.x, s * v.y, s * v.z, s * v.w);
}

__global__ void k_scatter(const float* __restrict__ h, const float* __restrict__ dinv,
                          const int* __restrict__ src, const int* __restrict__ dst,
                          float* __restrict__ tmp, int E) {
    long long gid = (long long)blockIdx.x * blockDim.x + threadIdx.x;
    if (gid >= (long long)E * 96) return;
    int e = (int)(gid / 96), c = (int)(gid % 96);
    int ns = src[e], nd = dst[e];
    float norm = dinv[ns] * dinv[nd];
    float4 v = ((const float4*)h)[(size_t)ns * 96 + c];
    float* o = tmp + (size_t)nd * D1 + c * 4;
    atomicAdd(o + 0, norm * v.x);
    atomicAdd(o + 1, norm * v.y);
    atomicAdd(o + 2, norm * v.z);
    atomicAdd(o + 3, norm * v.w);
}

// ---------------------------------------------------------------------------
// Epilogue 1: y_bf = bf16(lrelu(concat[x, gcn1 + b1]))  (N x 768 bf16, b128)
// ---------------------------------------------------------------------------
__global__ void k_epi1(const float* __restrict__ x, const float* __restrict__ tmp,
                       const float* __restrict__ b1, unsigned short* __restrict__ ybf,
                       int total8) {                                // N*96
    int i = blockIdx.x * blockDim.x + threadIdx.x;
    if (i >= total8) return;
    int node = i / 96, j = i % 96, d0 = j * 8;
    float4 a, b;
    if (d0 < D1) {
        const float4* p = (const float4*)(x + (size_t)node * D1 + d0);
        a = p[0]; b = p[1];
    } else {
        int dd = d0 - D1;
        const float4* p  = (const float4*)(tmp + (size_t)node * D1 + dd);
        const float4* bb = (const float4*)(b1 + dd);
        float4 t0 = p[0], t1 = p[1], c0 = bb[0], c1 = bb[1];
        a = make_float4(t0.x + c0.x, t0.y + c0.y, t0.z + c0.z, t0.w + c0.w);
        b = make_float4(t1.x + c1.x, t1.y + c1.y, t1.z + c1.z, t1.w + c1.w);
    }
    uint4 u = make_uint4(pk2(lrelu(a.x), lrelu(a.y)), pk2(lrelu(a.z), lrelu(a.w)),
                         pk2(lrelu(b.x), lrelu(b.y)), pk2(lrelu(b.z), lrelu(b.w)));
    ((uint4*)ybf)[i] = u;
}

// Epilogue 2: z = lrelu(tmp + b2) -> d_out (float4)
__global__ void k_epi2(const float* __restrict__ tmp, const float* __restrict__ b2,
                       float* __restrict__ z, int total4) {         // N*96
    int i = blockIdx.x * blockDim.x + threadIdx.x;
    if (i >= total4) return;
    int d = (i % 96) * 4;
    float4 v  = ((const float4*)tmp)[i];
    float4 bb = *(const float4*)(b2 + d);
    ((float4*)z)[i] = make_float4(lrelu(v.x + bb.x), lrelu(v.y + bb.y),
                                  lrelu(v.z + bb.z), lrelu(v.w + bb.w));
}

// Pooler: a[i] = z[i,:] . Wp + bp   (one wave32 per node)
__global__ void k_pool(const float* __restrict__ z, const float* __restrict__ Wp,
                       const float* __restrict__ bp, float* __restrict__ a, int N) {
    int w    = (int)((blockIdx.x * (long long)blockDim.x + threadIdx.x) >> 5);
    int lane = threadIdx.x & 31;
    if (w >= N) return;
    float s = 0.0f;
    for (int d = lane; d < D1; d += 32) s += z[(size_t)w * D1 + d] * Wp[d];
    #pragma unroll
    for (int off = 16; off >= 1; off >>= 1) s += __shfl_xor(s, off, 32);
    if (lane == 0) a[w] = s + bp[0];
}

// ---------------------------------------------------------------------------
extern "C" void kernel_launch(void* const* d_in, const int* in_sizes, int n_in,
                              void* d_out, int out_size, void* d_ws, size_t ws_size,
                              hipStream_t stream) {
    const float* x  = (const float*)d_in[0];
    const int*   ei = (const int*)  d_in[1];
    const float* W1 = (const float*)d_in[2];
    const float* b1 = (const float*)d_in[3];
    const float* W2 = (const float*)d_in[4];
    const float* b2 = (const float*)d_in[5];
    const float* Wp = (const float*)d_in[6];
    const float* bp = (const float*)d_in[7];

    const int N = in_sizes[0] / D1;
    const int E = in_sizes[1] / 2;
    const int* src = ei;
    const int* dst = ei + E;

    auto align256 = [](size_t v) { return (v + 255) & ~(size_t)255; };
    char* w = (char*)d_ws;
    size_t off = 0;
    float* dinv = (float*)(w + off);          off = align256(off + (size_t)N * 4);
    float* h    = (float*)(w + off);          off = align256(off + (size_t)N * D1 * 4);
    float* tmp  = (float*)(w + off);          off = align256(off + (size_t)N * D1 * 4);
    unsigned short* ybf  = (unsigned short*)(w + off); off = align256(off + (size_t)N * D2 * 2);
    unsigned short* Wsw1 = (unsigned short*)(w + off); off = align256(off + (size_t)D1 * D1 * 2);
    unsigned short* Wsw2 = (unsigned short*)(w + off); off = align256(off + (size_t)D2 * D1 * 2);
    (void)ws_size;

    float* z_out = (float*)d_out;             // N*384
    float* a_out = z_out + (size_t)N * D1;    // N*1

    const int B = 256;
    // 1. normalization
    k_fill1<<<(N + B - 1) / B, B, 0, stream>>>(dinv, N);
    k_deg  <<<(E + B - 1) / B, B, 0, stream>>>(dst, dinv, E);
    k_rsqrt<<<(N + B - 1) / B, B, 0, stream>>>(dinv, N);
    // 2. weights -> bf16 in WMMA fragment layout
    k_wswz<<<(D1 * D1 + B - 1) / B, B, 0, stream>>>(W1, Wsw1, D1);
    k_wswz<<<(D2 * D1 + B - 1) / B, B, 0, stream>>>(W2, Wsw2, D2);

    dim3 gg((N + 127) / 128, D1 / 64);
    size_t sh1 = (size_t)(D1 / 32) * 4096 + 8192;   // 56 KB
    size_t sh2 = (size_t)(D2 / 32) * 4096 + 8192;   // 104 KB
    long long sc = (long long)E * 96;
    int scGrid = (int)((sc + B - 1) / B);

    // 3. layer 1
    k_gemm_bf16<<<gg, B, sh1, stream>>>(x, 0, D1, Wsw1, h, N, D1);
    k_selfinit <<<(N * 96 + B - 1) / B, B, 0, stream>>>(h, dinv, tmp, N * 96);
    k_scatter  <<<scGrid, B, 0, stream>>>(h, dinv, src, dst, tmp, E);
    k_epi1     <<<(N * 96 + B - 1) / B, B, 0, stream>>>(x, tmp, b1, ybf, N * 96);

    // 4. layer 2
    k_gemm_bf16<<<gg, B, sh2, stream>>>(ybf, 1, D2, Wsw2, h, N, D2);
    k_selfinit <<<(N * 96 + B - 1) / B, B, 0, stream>>>(h, dinv, tmp, N * 96);
    k_scatter  <<<scGrid, B, 0, stream>>>(h, dinv, src, dst, tmp, E);
    k_epi2     <<<(N * 96 + B - 1) / B, B, 0, stream>>>(tmp, b2, z_out, N * 96);

    // 5. pooler
    k_pool<<<((N * 32) + B - 1) / B, B, 0, stream>>>(z_out, Wp, bp, a_out, N);
}
